// LrlcClfDebug_20822001451749
// MI455X (gfx1250) — matrix operand order repaired and checked by reading, not compiled, not run on hardware
//
#include <hip/hip_runtime.h>

typedef __attribute__((ext_vector_type(16))) _Float16 v16h;
typedef __attribute__((ext_vector_type(8)))  _Float16 v8h;
typedef __attribute__((ext_vector_type(8)))  float    v8f;

__device__ __forceinline__ v8f wmma_f32_f16(v16h a, v16h b, v8f c) {
  // (neg_a, A, neg_b, B, c_mod, C, reuse_a, reuse_b)
  return __builtin_amdgcn_wmma_f32_16x16x32_f16(false, a, false, b, (short)0, c, false, false);
}

template<int CIN, int COUT, int KS>
struct Cfg {
  static constexpr int KTOT   = CIN * KS * KS;
  static constexpr int KSTEPS = (KTOT + 31) / 32;
  static constexpr int KPAD   = KSTEPS * 32;
  static constexpr int COUTP  = ((COUT + 15) / 16) * 16;
  static constexpr int NT     = COUTP / 16;
  static constexpr int P      = KS / 2;
  static constexpr int PW     = 16 + KS - 1;          // patch width (16 pixels + halo)
  static constexpr int PATCH  = CIN * KS * PW;        // per-wave input patch elems
};

// A fragment: two contiguous 16B LDS loads (k: {kb..kb+7} U {kb+8..kb+15})
__device__ __forceinline__ v16h load_a(const _Float16* arow, int kb) {
  v8h lo = *(const v8h*)(arow + kb);
  v8h hi = *(const v8h*)(arow + kb + 8);
  return __builtin_shufflevector(lo, hi, 0,1,2,3,4,5,6,7,8,9,10,11,12,13,14,15);
}

// ---- staging helpers ----

// weights transposed to [n][k] (k contiguous, zero padded)
template<int CIN, int COUT, int KS>
__device__ void stage_weights_t(const float* __restrict__ w, _Float16* smWt) {
  using C = Cfg<CIN, COUT, KS>;
  for (int idx = threadIdx.x; idx < C::COUTP * C::KPAD; idx += blockDim.x) {
    int n = idx / C::KPAD, k = idx - n * C::KPAD;
    _Float16 v = (_Float16)0.f;
    if (k < C::KTOT && n < COUT) {
      int ic  = k / (KS * KS);
      int rem = k % (KS * KS);                  // == kh*KS + kw (OIHW)
      v = (_Float16)w[(n * CIN + ic) * (KS * KS) + rem];
    }
    smWt[idx] = v;
  }
}

template<int CIN, int COUT, int KS>
__device__ void stage_patch(const float* __restrict__ in, _Float16* smP,
                            int n, int h, int w0, int H, int W, int lane, bool active) {
  using C = Cfg<CIN, COUT, KS>;
  if (!active) return;
  for (int idx = lane; idx < C::PATCH; idx += 32) {
    int ic = idx / (KS * C::PW);
    int r  = (idx / C::PW) % KS;
    int c  = idx % C::PW;
    int ih = h + r - C::P;
    int iw = w0 + c - C::P;
    float v = 0.f;
    if (ih >= 0 && ih < H && iw >= 0 && iw < W)
      v = in[((n * CIN + ic) * H + ih) * W + iw];
    smP[idx] = (_Float16)v;
  }
}

// expand patch -> per-wave im2col tile smA[m][k] (k contiguous, zero padded)
template<int KS, int PW, int KTOT, int KPAD>
__device__ void expand_im2col(const _Float16* smP, _Float16* smA, int lane, bool active) {
  if (!active) return;
  for (int t = lane; t < 16 * KPAD; t += 32) {
    int m = t / KPAD, k = t - m * KPAD;
    _Float16 v = (_Float16)0.f;
    if (k < KTOT) {
      int ic  = k / (KS * KS);
      int rem = k % (KS * KS);
      int kh  = rem / KS;
      int kw  = rem % KS;
      v = smP[(ic * KS + kh) * PW + (m + kw)];
    }
    smA[t] = v;          // t == m*KPAD + k
  }
}

// ---- pass 1: conv (no store) + per-channel sum / sumsq for BN ----

template<int CIN, int COUT, int KS, int WPB>
__global__ __launch_bounds__(WPB * 32)
void conv_stats_kernel(const float* __restrict__ in, const float* __restrict__ w,
                       float* __restrict__ stats, int Bn, int H, int W) {
  using C = Cfg<CIN, COUT, KS>;
  __shared__ alignas(32) _Float16 smWt[C::COUTP * C::KPAD];
  __shared__ alignas(32) _Float16 smP[WPB][C::PATCH];
  __shared__ alignas(32) _Float16 smA[WPB][16 * C::KPAD];
  __shared__ float smS[COUT * 2];
  const int tid = threadIdx.x, lane = tid & 31, wid = tid >> 5;
  const int wt = W >> 4;
  const long tasks = (long)Bn * H * wt;
  const long gw = (long)blockIdx.x * WPB + wid;
  const bool active = gw < tasks;
  int w0 = 0, h = 0, n = 0;
  if (active) {
    w0 = (int)(gw % wt) * 16;
    long t = gw / wt;
    h = (int)(t % H);
    n = (int)(t / H);
  }
  stage_weights_t<CIN, COUT, KS>(w, smWt);
  stage_patch<CIN, COUT, KS>(in, smP[wid], n, h, w0, H, W, lane, active);
  expand_im2col<KS, C::PW, C::KTOT, C::KPAD>(smP[wid], smA[wid], lane, active);
  if (tid < COUT * 2) smS[tid] = 0.f;
  __syncthreads();
  if (active) {
    const _Float16* arow = smA[wid] + (lane & 15) * C::KPAD;
    const int hi4 = lane >> 4;
#pragma unroll
    for (int nt = 0; nt < C::NT; ++nt) {
      const _Float16* brow = smWt + ((lane & 15) + nt * 16) * C::KPAD;
      v8f acc = {};
      for (int s = 0; s < C::KSTEPS; ++s) {
        v16h a = load_a(arow, s * 32 + hi4 * 16);
        v16h b = *(const v16h*)(brow + s * 32 + hi4 * 16);
        acc = wmma_f32_f16(a, b, acc);
      }
      int oc = (lane & 15) + nt * 16;
      if (oc < COUT) {
        float s1 = 0.f, s2 = 0.f;
#pragma unroll
        for (int r = 0; r < 8; ++r) { float v = acc[r]; s1 += v; s2 += v * v; }
        atomicAdd(&smS[oc * 2 + 0], s1);
        atomicAdd(&smS[oc * 2 + 1], s2);
      }
    }
  }
  __syncthreads();
  if (tid < COUT * 2) atomicAdd(&stats[tid], smS[tid]);
}

// ---- pass 2: recompute conv, fuse BN-normalize + ReLU + 2x2 maxpool ----

template<int CIN, int COUT, int KS, int WPB>
__global__ __launch_bounds__(WPB * 32)
void conv_bn_pool_kernel(const float* __restrict__ in, const float* __restrict__ w,
                         const float* __restrict__ stats, const float* __restrict__ g,
                         const float* __restrict__ bb, float* __restrict__ out,
                         int Bn, int H, int W, float invCnt) {
  using C = Cfg<CIN, COUT, KS>;
  __shared__ alignas(32) _Float16 smWt[C::COUTP * C::KPAD];
  __shared__ alignas(32) _Float16 smP[WPB][C::PATCH];
  __shared__ alignas(32) _Float16 smA[WPB][16 * C::KPAD];
  __shared__ float smO[WPB][C::COUTP][16];
  __shared__ float smScale[COUT], smShift[COUT];
  const int tid = threadIdx.x, lane = tid & 31, wid = tid >> 5;
  const int wt = W >> 4, hb = H >> 2;
  long task = blockIdx.x;                 // grid == Bn*hb*wt exactly
  const int w0 = (int)(task % wt) * 16;
  long t = task / wt;
  const int h0 = (int)(t % hb) * 4;
  const int n  = (int)(t / hb);
  const int h  = h0 + wid;                // wave -> one full-res row
  stage_weights_t<CIN, COUT, KS>(w, smWt);
  stage_patch<CIN, COUT, KS>(in, smP[wid], n, h, w0, H, W, lane, true);
  expand_im2col<KS, C::PW, C::KTOT, C::KPAD>(smP[wid], smA[wid], lane, true);
  if (tid < COUT) {
    float s1 = stats[tid * 2], s2 = stats[tid * 2 + 1];
    float mean = s1 * invCnt;
    float var  = fmaxf(s2 * invCnt - mean * mean, 0.f);
    float sc   = g[tid] * rsqrtf(var + 1e-5f);
    smScale[tid] = sc;
    smShift[tid] = bb[tid] - mean * sc;   // conv bias cancels in train-mode BN
  }
  __syncthreads();
  const _Float16* arow = smA[wid] + (lane & 15) * C::KPAD;
  const int hi4 = lane >> 4;
#pragma unroll
  for (int nt = 0; nt < C::NT; ++nt) {
    const _Float16* brow = smWt + ((lane & 15) + nt * 16) * C::KPAD;
    v8f acc = {};
    for (int s = 0; s < C::KSTEPS; ++s) {
      v16h a = load_a(arow, s * 32 + hi4 * 16);
      v16h b = *(const v16h*)(brow + s * 32 + hi4 * 16);
      acc = wmma_f32_f16(a, b, acc);
    }
    int oc = (lane & 15) + nt * 16;
    if (oc < COUT) {
#pragma unroll
      for (int r = 0; r < 8; ++r) {
        int m = r + 8 * hi4;
        smO[wid][oc][m] = fmaxf(acc[r] * smScale[oc] + smShift[oc], 0.f);
      }
    }
  }
  __syncthreads();
  const int HP = H >> 1, WP = W >> 1;
  for (int idx = tid; idx < COUT * 16; idx += (WPB * 32)) {
    int pr  = idx / (COUT * 8);
    int rem = idx % (COUT * 8);
    int oc  = rem / 8, pc = rem % 8;
    int r0 = pr * 2, c0 = pc * 2;
    float v = fmaxf(fmaxf(smO[r0][oc][c0], smO[r0][oc][c0 + 1]),
                    fmaxf(smO[r0 + 1][oc][c0], smO[r0 + 1][oc][c0 + 1]));
    out[((n * COUT + oc) * HP + (h0 >> 1) + pr) * WP + (w0 >> 1) + pc] = v;
  }
}

// ---- combining-weight softmax: cw[r,h,w] = softmax_r(cw_h[r,h] + cw_w[r,w]) ----

__global__ void cw_kernel(const float* __restrict__ ch, const float* __restrict__ cwv,
                          float* __restrict__ out) {
  int idx = blockIdx.x * blockDim.x + threadIdx.x;
  if (idx >= 1024) return;
  int h = idx >> 5, w = idx & 31;
  float e[4], mx = -1e30f;
#pragma unroll
  for (int r = 0; r < 4; ++r) { e[r] = ch[r * 32 + h] + cwv[r * 32 + w]; mx = fmaxf(mx, e[r]); }
  float s = 0.f;
#pragma unroll
  for (int r = 0; r < 4; ++r) { e[r] = expf(e[r] - mx); s += e[r]; }
  float inv = 1.f / s;
#pragma unroll
  for (int r = 0; r < 4; ++r) out[(r * 32 + h) * 32 + w] = e[r] * inv;
}

// ---- LRLC: rank-basis conv (WMMA) + per-pixel rank combine + bias + ReLU ----
// 2 waves/block (LDS budget); im2col built directly from global (L2-resident).
// Output written as f16 for the fc1 WMMA GEMM.

__global__ __launch_bounds__(64)
void lrlc_kernel(const float* __restrict__ in, const float* __restrict__ w,
                 const float* __restrict__ cw, const float* __restrict__ lb,
                 _Float16* __restrict__ out, int Bn) {
  using C = Cfg<32, 32, 3>;
  __shared__ alignas(32) _Float16 smWt[C::COUTP * C::KPAD];   // 18KB
  __shared__ alignas(32) _Float16 smA[2][16 * C::KPAD];       // 18KB
  __shared__ float smY[2][32][16];                            // 4KB
  const int tid = threadIdx.x, lane = tid & 31, wid = tid >> 5;
  const int H = 32, W = 32, wt = 2;
  const long tasks = (long)Bn * H * wt;
  const long gw = (long)blockIdx.x * 2 + wid;
  const bool active = gw < tasks;
  int w0 = 0, h = 0, n = 0;
  if (active) {
    w0 = (int)(gw % wt) * 16;
    long t = gw / wt;
    h = (int)(t % H);
    n = (int)(t / H);
  }
  stage_weights_t<32, 32, 3>(w, smWt);
  if (active) {
    for (int t = lane; t < 16 * C::KPAD; t += 32) {
      int m = t / C::KPAD, k = t - m * C::KPAD;   // KPAD == KTOT == 288
      int ic  = k / 9, rem = k % 9;
      int kh  = rem / 3, kw = rem % 3;
      int ih  = h + kh - 1, iw = w0 + m + kw - 1;
      float v = 0.f;
      if (ih >= 0 && ih < H && iw >= 0 && iw < W)
        v = in[((n * 32 + ic) * H + ih) * W + iw];
      smA[wid][t] = (_Float16)v;
    }
  }
  __syncthreads();
  if (active) {
    const _Float16* arow = smA[wid] + (lane & 15) * C::KPAD;
    const int hi4 = lane >> 4;
#pragma unroll
    for (int nt = 0; nt < 2; ++nt) {
      const _Float16* brow = smWt + ((lane & 15) + nt * 16) * C::KPAD;
      v8f acc = {};
      for (int s = 0; s < C::KSTEPS; ++s) {
        v16h a = load_a(arow, s * 32 + hi4 * 16);
        v16h b = *(const v16h*)(brow + s * 32 + hi4 * 16);
        acc = wmma_f32_f16(a, b, acc);
      }
      int oc = (lane & 15) + nt * 16;
#pragma unroll
      for (int r = 0; r < 8; ++r) smY[wid][oc][r + 8 * hi4] = acc[r];
    }
  }
  __syncthreads();
  if (active) {
    for (int idx = lane; idx < 8 * 16; idx += 32) {
      int c = idx >> 4, m = idx & 15;
      int ww = w0 + m;
      float v = lb[c];
#pragma unroll
      for (int r = 0; r < 4; ++r)
        v += smY[wid][r * 8 + c][m] * cw[(r * 32 + h) * 32 + ww];
      out[((n * 8 + c) * 32 + h) * 32 + ww] = (_Float16)fmaxf(v, 0.f);
    }
  }
}

// ---- fc1: [128,8192](f16) x [8192,64](f16) WMMA GEMM + bias + ReLU ----

__global__ __launch_bounds__(128)
void fc1_kernel(const _Float16* __restrict__ x, const _Float16* __restrict__ w,
                const float* __restrict__ bias, float* __restrict__ out) {
  const int lane = threadIdx.x & 31, wid = threadIdx.x >> 5;
  const int task = blockIdx.x * 4 + wid;       // 32 tasks: 8 mtiles x 4 ntiles
  const int mt = task >> 2, nt = task & 3;
  const int hi4 = lane >> 4;
  const _Float16* arow = x + (mt * 16 + (lane & 15)) * 8192;
  const _Float16* brow = w + (nt * 16 + (lane & 15)) * 8192;
  v8f acc = {};
  for (int s = 0; s < 256; ++s) {
    v16h a = load_a(arow, s * 32 + hi4 * 16);            // 2x16B global loads
    v16h b = *(const v16h*)(brow + s * 32 + hi4 * 16);   // 32B global load
    acc = wmma_f32_f16(a, b, acc);
  }
  const int col = nt * 16 + (lane & 15);
#pragma unroll
  for (int r = 0; r < 8; ++r) {
    int mm = mt * 16 + r + 8 * hi4;
    out[mm * 64 + col] = fmaxf(acc[r] + bias[col], 0.f);
  }
}

// ---- fc2: [128,64] x [64,10] + bias (tiny, VALU) ----

__global__ void fc2_kernel(const float* __restrict__ f, const float* __restrict__ w,
                           const float* __restrict__ b, float* __restrict__ out) {
  int idx = blockIdx.x * blockDim.x + threadIdx.x;
  if (idx >= 128 * 10) return;
  int i = idx / 10, o = idx % 10;
  float acc = b[o];
#pragma unroll 8
  for (int k = 0; k < 64; ++k) acc += f[i * 64 + k] * w[o * 64 + k];
  out[idx] = acc;
}

__global__ void cvt_f16_kernel(const float* __restrict__ in, _Float16* __restrict__ out, int n) {
  int i = blockIdx.x * blockDim.x + threadIdx.x;
  if (i < n) out[i] = (_Float16)in[i];
}

__global__ void zero_kernel(float* __restrict__ p, int n) {
  int i = blockIdx.x * blockDim.x + threadIdx.x;
  if (i < n) p[i] = 0.f;
}

extern "C" void kernel_launch(void* const* d_in, const int* in_sizes, int n_in,
                              void* d_out, int out_size, void* d_ws, size_t ws_size,
                              hipStream_t stream) {
  (void)n_in; (void)out_size; (void)ws_size;
  const float* x     = (const float*)d_in[0];
  const float* c1w   = (const float*)d_in[1];
  const float* bn1g  = (const float*)d_in[3];
  const float* bn1b  = (const float*)d_in[4];
  const float* c2w   = (const float*)d_in[5];
  const float* bn2g  = (const float*)d_in[7];
  const float* bn2b  = (const float*)d_in[8];
  const float* c3w   = (const float*)d_in[9];
  const float* bn3g  = (const float*)d_in[11];
  const float* bn3b  = (const float*)d_in[12];
  const float* lrlcw = (const float*)d_in[13];
  const float* lrlcb = (const float*)d_in[14];
  const float* cwh   = (const float*)d_in[15];
  const float* cww   = (const float*)d_in[16];
  const float* fc1w  = (const float*)d_in[17];
  const float* fc1b  = (const float*)d_in[18];
  const float* fc2w  = (const float*)d_in[19];
  const float* fc2b  = (const float*)d_in[20];
  float* out = (float*)d_out;

  const int B = in_sizes[0] / (256 * 256);   // 128

  // workspace layout (float units; all offsets multiples of 64 -> 256B aligned)
  float* ws    = (float*)d_ws;
  float* stats = ws;                                   // 256
  float* cwbuf = ws + 256;                             // 4096
  float* fbuf  = cwbuf + 4096;                         // 8192
  float* h3    = fbuf + 8192;                          // B*32*32*32
  float* h2    = h3 + (long)B * 32 * 32 * 32;          // B*16*64*64
  float* h1    = h2 + (long)B * 16 * 64 * 64;          // B*8*128*128
  _Float16* ych   = (_Float16*)(h1 + (long)B * 8 * 128 * 128);   // B*8192 halves
  _Float16* fc1wh = (_Float16*)((float*)ych + (long)B * 4096);   // 64*8192 halves

  zero_kernel<<<1, 256, 0, stream>>>(stats, 256);
  cw_kernel<<<4, 256, 0, stream>>>(cwh, cww, cwbuf);
  cvt_f16_kernel<<<2048, 256, 0, stream>>>(fc1w, fc1wh, 64 * 8192);

  // layer 1: 1->8, 5x5, 256x256 -> pool -> 128x128
  {
    unsigned sblocks = (unsigned)(((long)B * 256 * (256 / 16) + 3) / 4);
    conv_stats_kernel<1, 8, 5, 4><<<sblocks, 128, 0, stream>>>(x, c1w, stats + 0, B, 256, 256);
    unsigned pblocks = (unsigned)((long)B * (256 / 4) * (256 / 16));
    conv_bn_pool_kernel<1, 8, 5, 4><<<pblocks, 128, 0, stream>>>(
        x, c1w, stats + 0, bn1g, bn1b, h1, B, 256, 256, 1.f / ((float)B * 256 * 256));
  }
  // layer 2: 8->16, 5x5, 128x128 -> pool -> 64x64
  {
    unsigned sblocks = (unsigned)(((long)B * 128 * (128 / 16) + 3) / 4);
    conv_stats_kernel<8, 16, 5, 4><<<sblocks, 128, 0, stream>>>(h1, c2w, stats + 64, B, 128, 128);
    unsigned pblocks = (unsigned)((long)B * (128 / 4) * (128 / 16));
    conv_bn_pool_kernel<8, 16, 5, 4><<<pblocks, 128, 0, stream>>>(
        h1, c2w, stats + 64, bn2g, bn2b, h2, B, 128, 128, 1.f / ((float)B * 128 * 128));
  }
  // layer 3: 16->32, 3x3, 64x64 -> pool -> 32x32
  {
    unsigned sblocks = (unsigned)(((long)B * 64 * (64 / 16) + 3) / 4);
    conv_stats_kernel<16, 32, 3, 4><<<sblocks, 128, 0, stream>>>(h2, c3w, stats + 128, B, 64, 64);
    unsigned pblocks = (unsigned)((long)B * (64 / 4) * (64 / 16));
    conv_bn_pool_kernel<16, 32, 3, 4><<<pblocks, 128, 0, stream>>>(
        h2, c3w, stats + 128, bn3g, bn3b, h3, B, 64, 64, 1.f / ((float)B * 64 * 64));
  }
  // LRLC: 32->(4 rank x 8 ch), 3x3 @ 32x32, rank-combined + bias + ReLU -> f16
  {
    unsigned blocks = (unsigned)(((long)B * 32 * 2 + 1) / 2);
    lrlc_kernel<<<blocks, 64, 0, stream>>>(h3, lrlcw, cwbuf, lrlcb, ych, B);
  }
  // FC head
  fc1_kernel<<<8, 128, 0, stream>>>(ych, fc1wh, fc1b, fbuf);
  fc2_kernel<<<5, 256, 0, stream>>>(fbuf, fc2w, fc2b, out);
}
